// ContrastiveCenterLoss_32289564131921
// MI455X (gfx1250) — matrix operand order repaired and checked by reading, not compiled, not run on hardware
//
#include <hip/hip_runtime.h>

typedef __attribute__((ext_vector_type(16))) __bf16 v16bf;
typedef __attribute__((ext_vector_type(8)))  __bf16 v8bf;
typedef __attribute__((ext_vector_type(8)))  float  v8f;

#define D_DIM   2048
#define C_DIM   1000
#define C_PAD   1008
#define NTILES  63        // C_PAD / 16
#define WAVES   8
#define B_DIM   32768
#define ROWS    32        // rows per block (two 16-row WMMA tiles)
#define NBLOCKS (B_DIM / ROWS)
#define INV_T   (1.0f / 0.07f)

__device__ __forceinline__ unsigned short f32_to_bf16_rne(float x) {
    unsigned int u = __builtin_bit_cast(unsigned int, x);
    unsigned int r = (u + 0x7FFFu + ((u >> 16) & 1u)) >> 16;
    return (unsigned short)r;
}

// ---------------- Kernel 1: normalize centers -> bf16, zero-pad to 1008 rows
__global__ __launch_bounds__(256) void cc_centers(const float* __restrict__ centers,
                                                  unsigned short* __restrict__ cbf) {
    const int c = blockIdx.x;
    const int t = threadIdx.x;
    __shared__ float red[256];
    if (c < C_DIM) {
        const float* row = centers + (size_t)c * D_DIM;
        float ssq = 0.f;
        for (int k = t; k < D_DIM; k += 256) { float x = row[k]; ssq += x * x; }
        red[t] = ssq;
        __syncthreads();
        for (int off = 128; off > 0; off >>= 1) {
            if (t < off) red[t] += red[t + off];
            __syncthreads();
        }
        const float inv = 1.0f / fmaxf(sqrtf(red[0]), 1e-12f);
        for (int k = t; k < D_DIM; k += 256)
            cbf[(size_t)c * D_DIM + k] = f32_to_bf16_rne(row[k] * inv);
    } else {
        for (int k = t; k < D_DIM; k += 256)
            cbf[(size_t)c * D_DIM + k] = 0;
    }
}

// ---------------- Kernel 2: fused normalize + bf16 WMMA GEMM + logsumexp
// 32 rows per block, 8 waves; each wave: 2 row-tiles x (63/8) col-tiles.
// No max-shift needed: |sim|<=1 -> sim/T in [-14.3,14.3], exp() is f32-safe.
__global__ __launch_bounds__(256) void cc_main(const float* __restrict__ feats,
                                               const int* __restrict__ targets,
                                               const unsigned short* __restrict__ cbf,
                                               float* __restrict__ partials) {
    __shared__ unsigned short aTile[ROWS * D_DIM];   // 128 KB bf16 A tile
    __shared__ float2 sRed[WAVES * 32 * 16];         // 32 KB: (sumexp, pos) per wave/lane/rowslot
    __shared__ float sNorm[ROWS * 8];
    __shared__ float sPer[ROWS];
    __shared__ int   sTgt[ROWS];

    const int t = threadIdx.x;
    const int rowBase = blockIdx.x * ROWS;

    // ---- Phase 1: per-row sum of squares (8 threads per row, 256 elems each)
    const int r = t >> 3;     // row 0..31
    const int s = t & 7;      // sub-slice 0..7
    const float* frow = feats + (size_t)(rowBase + r) * D_DIM;
    float ssq = 0.f;
    for (int k = s * 256; k < s * 256 + 256; k += 4) {
        float4 v = *(const float4*)(frow + k);
        ssq += v.x * v.x + v.y * v.y + v.z * v.z + v.w * v.w;
    }
    sNorm[r * 8 + s] = ssq;
    if (t < ROWS) sTgt[t] = targets[rowBase + t];
    __syncthreads();

    float tot = 0.f;
    #pragma unroll
    for (int i = 0; i < 8; ++i) tot += sNorm[r * 8 + i];
    const float inv = 1.0f / fmaxf(sqrtf(tot), 1e-12f);
    for (int k = s * 256; k < s * 256 + 256; ++k)
        aTile[r * D_DIM + k] = f32_to_bf16_rne(frow[k] * inv);
    __syncthreads();

    // ---- Phase 2: WMMA GEMM over column tiles, fused exp-sum + target pick
    const int wave = t >> 5;
    const int lane = t & 31;
    const int half = lane >> 4;   // K-half selector: k in [16h, 16h+16) of each 32-k step (contiguous order)
    const int ln16 = lane & 15;   // A: row m ; B: col n ; C/D: col n

    float sm0[8], sm1[8], ps0[8], ps1[8];
    int   tg0[8], tg1[8];
    #pragma unroll
    for (int i = 0; i < 8; ++i) {
        sm0[i] = 0.f; sm1[i] = 0.f; ps0[i] = 0.f; ps1[i] = 0.f;
        tg0[i] = sTgt[i + 8 * half];
        tg1[i] = sTgt[16 + i + 8 * half];
    }

    const unsigned short* arow0 = aTile + ln16 * D_DIM + 16 * half;          // rows 0..15
    const unsigned short* arow1 = aTile + (16 + ln16) * D_DIM + 16 * half;   // rows 16..31

    for (int ct = wave; ct < NTILES; ct += WAVES) {
        v8f acc0 = {0.f, 0.f, 0.f, 0.f, 0.f, 0.f, 0.f, 0.f};
        v8f acc1 = {0.f, 0.f, 0.f, 0.f, 0.f, 0.f, 0.f, 0.f};
        const unsigned short* bcol = cbf + (size_t)(ct * 16 + ln16) * D_DIM + 16 * half;
        #pragma unroll 4
        for (int k0 = 0; k0 < D_DIM; k0 += 32) {
            // B frag: contiguous K [k0+16h, +16) of center row (ct*16 + n)
            v16bf b = *(const v16bf*)(bcol + k0);
            // A frags: K chunks [k0+16h, +8) and [k0+16h+8, +8)  (contiguous K, matches B)
            v8bf a0lo = *(const v8bf*)(arow0 + k0);
            v8bf a0hi = *(const v8bf*)(arow0 + k0 + 8);
            v8bf a1lo = *(const v8bf*)(arow1 + k0);
            v8bf a1hi = *(const v8bf*)(arow1 + k0 + 8);
            v16bf a0 = __builtin_shufflevector(a0lo, a0hi,
                         0, 1, 2, 3, 4, 5, 6, 7, 8, 9, 10, 11, 12, 13, 14, 15);
            v16bf a1 = __builtin_shufflevector(a1lo, a1hi,
                         0, 1, 2, 3, 4, 5, 6, 7, 8, 9, 10, 11, 12, 13, 14, 15);
            acc0 = __builtin_amdgcn_wmma_f32_16x16x32_bf16(
                       false, a0, false, b, (short)0, acc0, false, false);
            acc1 = __builtin_amdgcn_wmma_f32_16x16x32_bf16(
                       false, a1, false, b, (short)0, acc1, false, false);
        }
        const int c = ct * 16 + ln16;
        // Branchless epilogue: exp computed unconditionally, gated by 0/1 mask
        // via fmaf so the compiler emits v_cndmask + v_fmac, not saveexec blocks.
        const float mval = (c < C_DIM) ? 1.0f : 0.0f;
        #pragma unroll
        for (int i = 0; i < 8; ++i) {
            const float v0 = acc0[i] * INV_T;
            const float v1 = acc1[i] * INV_T;
            sm0[i] = fmaf(mval, __expf(v0), sm0[i]);
            sm1[i] = fmaf(mval, __expf(v1), sm1[i]);
            ps0[i] = fmaf((c == tg0[i]) ? mval : 0.0f, v0, ps0[i]);
            ps1[i] = fmaf((c == tg1[i]) ? mval : 0.0f, v1, ps1[i]);
        }
    }

    // per-lane row-slot j: tile0 -> i, tile1 -> 8+i  (row m = 8*half + i [+16])
    {
        float2* dst = sRed + (wave * 32 + lane) * 16;
        #pragma unroll
        for (int i = 0; i < 8; ++i) {
            dst[i]     = make_float2(sm0[i], ps0[i]);
            dst[8 + i] = make_float2(sm1[i], ps1[i]);
        }
    }
    __syncthreads();

    // ---- Phase 3: per-row merge across 8 waves x 16 owning lanes (pure sums)
    if (t < ROWS) {
        const int m = t;
        const int lbase = ((m >> 3) & 1) * 16;        // which lane-half owns row m
        const int j = (m & 7) + 8 * (m >> 4);         // row-slot within lane
        float S = 0.f, P = 0.f;
        for (int w = 0; w < WAVES; ++w) {
            for (int i = 0; i < 16; ++i) {
                const float2 e = sRed[(w * 32 + lbase + i) * 16 + j];
                S += e.x;
                P += e.y;
            }
        }
        const float lse = __logf(S);
        const int tg = sTgt[m];
        const float sc = (tg == 1 || tg == 2) ? 2.0f : 1.0f;
        sPer[m] = (lse - P) * sc;
    }
    __syncthreads();
    if (t == 0) {
        float sum = 0.f;
        #pragma unroll
        for (int i = 0; i < ROWS; ++i) sum += sPer[i];
        partials[blockIdx.x] = sum;
    }
}

// ---------------- Kernel 3: reduce block partials -> mean
__global__ __launch_bounds__(256) void cc_final(const float* __restrict__ partials,
                                                float* __restrict__ out) {
    __shared__ float red[256];
    const int t = threadIdx.x;
    float sacc = 0.f;
    for (int i = t; i < NBLOCKS; i += 256) sacc += partials[i];
    red[t] = sacc;
    __syncthreads();
    for (int off = 128; off > 0; off >>= 1) {
        if (t < off) red[t] += red[t + off];
        __syncthreads();
    }
    if (t == 0) out[0] = red[0] * (1.0f / (float)B_DIM);   // LAMBDA = 1
}

extern "C" void kernel_launch(void* const* d_in, const int* in_sizes, int n_in,
                              void* d_out, int out_size, void* d_ws, size_t ws_size,
                              hipStream_t stream) {
    const float* feats   = (const float*)d_in[0];
    const int*   targets = (const int*)d_in[1];
    const float* centers = (const float*)d_in[2];
    float* out = (float*)d_out;

    unsigned short* cbf = (unsigned short*)d_ws;                       // 1008*2048*2 B
    float* partials = (float*)((char*)d_ws + (size_t)C_PAD * D_DIM * 2);

    cc_centers<<<C_PAD, 256, 0, stream>>>(centers, cbf);
    cc_main<<<NBLOCKS, 256, 0, stream>>>(feats, targets, cbf, partials);
    cc_final<<<1, 256, 0, stream>>>(partials, out);
}
